// QuantizedLinear_910533067246
// MI455X (gfx1250) — matrix operand order repaired and checked by reading, not compiled, not run on hardware
//
#include <hip/hip_runtime.h>

typedef __attribute__((ext_vector_type(16))) _Float16 v16h;
typedef __attribute__((ext_vector_type(8)))  _Float16 v8h;
typedef __attribute__((ext_vector_type(8)))  float    v8f;
typedef __attribute__((ext_vector_type(4)))  float    v4f;
typedef __attribute__((ext_vector_type(4)))  int      v4i;
typedef __attribute__((ext_vector_type(8)))  int      v8i;
typedef __attribute__((ext_vector_type(4)))  unsigned int v4u;

constexpr int kDIn  = 4096;
constexpr int kDOut = 16384;
constexpr int kRows = 8 * 2048;                        // B * S = 16384
constexpr long long kElems = (long long)kDOut * kDIn;  // == kRows * kDIn

constexpr int BM = 128;
constexpr int BN = 128;
constexpr int BK = 32;
constexpr int LDK = BK + 8;            // 64B data + 16B pad per row (80B stride)
constexpr int TILE_H = 128 * LDK;      // halves per padded 128x32 tile (10240 B)

#if defined(__has_builtin)
#if __has_builtin(__builtin_amdgcn_tensor_load_to_lds) && __has_builtin(__builtin_amdgcn_s_wait_tensorcnt)
#define USE_TDM 1
#endif
#endif

// ---------------- Pass 1: convert x -> f16, dequantize q -> f16 -------------
__global__ __launch_bounds__(256)
void cvt_dequant_kernel(const float* __restrict__ X,
                        const int*   __restrict__ Q,
                        const float* __restrict__ scale_p,
                        const float* __restrict__ zp_p,
                        _Float16* __restrict__ Xh,
                        _Float16* __restrict__ Wh)
{
    const float scale = scale_p[0];
    const float zp    = zp_p[0];
    const size_t i0 = ((size_t)blockIdx.x * 256 + threadIdx.x) * 8;

    v4f x0 = *(const v4f*)(X + i0);
    v4f x1 = *(const v4f*)(X + i0 + 4);
    v4i q0 = *(const v4i*)(Q + i0);
    v4i q1 = *(const v4i*)(Q + i0 + 4);

    v8h xh, wh;
#pragma unroll
    for (int c = 0; c < 4; ++c) {
        xh[c]     = (_Float16)x0[c];
        xh[4 + c] = (_Float16)x1[c];
        wh[c]     = (_Float16)(((float)q0[c] - zp) * scale);
        wh[4 + c] = (_Float16)(((float)q1[c] - zp) * scale);
    }
    *(v8h*)(Xh + i0) = xh;
    *(v8h*)(Wh + i0) = wh;
}

#ifdef USE_TDM
// Build D# group0: count=1 | lds_addr[63:32] | global_addr[120:64] | type=2
__device__ __forceinline__ v4u tdm_g0(unsigned lds_byte_off, const _Float16* gptr) {
    unsigned long long ga = (unsigned long long)(uintptr_t)gptr;
    v4u g;
    g[0] = 1u;                                   // count=1, user descriptor
    g[1] = lds_byte_off;                         // LDS dest (bytes)
    g[2] = (unsigned)ga;                         // global addr [31:0]
    g[3] = (unsigned)(ga >> 32) | (2u << 30);    // addr [56:32] | type=2
    return g;
}

// D# group1 for a 128-row x 32-elem f16 tile out of a [16384 x 4096] tensor,
// with LDS padding: 16 DWORDs (one 64B row) then 4 DWORDs (16B) pad -> 80B stride.
__device__ __forceinline__ v8i tdm_g1() {
    v8i g;
    g[0] = (1 << 16)      // data_size = 2 bytes
         | (1 << 20)      // pad_enable
         | (3 << 22)      // pad_interval: 16 DWORDs (one 64B tile row)
         | (3 << 25);     // pad_amount:   4 DWORDs (16B)
    g[1] = (int)(4096u << 16);   // tensor_dim0 = 4096 (low 16 bits at [63:48])
    g[2] = (int)(16384u << 16);  // dim0 hi = 0, tensor_dim1 = 16384 (low 16)
    g[3] = (int)(32u << 16);     // dim1 hi = 0, tile_dim0 = 32
    g[4] = 128;                  // tile_dim1 = 128, tile_dim2 = 0
    g[5] = 4096;                 // tensor_dim0_stride = 4096 elements
    g[6] = 0;
    g[7] = 0;
    return g;
}

__device__ __forceinline__ void tdm_load_tile(unsigned lds_byte_off,
                                              const _Float16* gptr, v8i g1) {
    v4i z4 = {0, 0, 0, 0};
    v8i z8 = {0, 0, 0, 0, 0, 0, 0, 0};
    __builtin_amdgcn_tensor_load_to_lds(tdm_g0(lds_byte_off, gptr), g1,
                                        z4, z4, z8, 0);
}
#endif

// ---------------- Pass 2: C = Xh * Wh^T + bias (f16 WMMA, f32 accum) --------
__global__ __launch_bounds__(256)
void qlinear_wmma_kernel(const _Float16* __restrict__ Xh,
                         const _Float16* __restrict__ Wh,
                         const float* __restrict__ bias,
                         float* __restrict__ Out)
{
    // [buf0: A,W][buf1: A,W], each tile 128 rows x 80B = 10240 B (40 KB total)
    __shared__ __align__(16) _Float16 sh[4 * TILE_H];

    const int n_blk = blockIdx.x * BN;   // x fastest over N: W streams, X slice hot in L2
    const int m_blk = blockIdx.y * BM;

    const int tid  = threadIdx.x;
    const int wave = tid >> 5;
    const int lane = tid & 31;
    const int fr   = lane & 15;
    const int hi   = lane >> 4;

    const int wm = (wave & 3) * 32;      // 4 waves along M, 32 rows each
    const int wn = (wave >> 2) * 64;     // 2 waves along N, 64 cols each

    const _Float16* Arow = Xh + (size_t)m_blk * kDIn;
    const _Float16* Wrow = Wh + (size_t)n_blk * kDIn;

    v8f acc[2][4] = {};
    constexpr int nT = kDIn / BK;        // 128 K-steps

#ifdef USE_TDM
    const v8i g1 = tdm_g1();
    if (wave == 0) {                     // preload buffer 0 via Tensor Data Mover
        tdm_load_tile(0u, Arow, g1);
        tdm_load_tile((unsigned)(TILE_H * sizeof(_Float16)), Wrow, g1);
    }
#endif

    for (int t = 0; t < nT; ++t) {
        const int buf = t & 1;
        const _Float16* Ab = sh + buf * (2 * TILE_H);
        const _Float16* Wb = Ab + TILE_H;

#ifdef USE_TDM
        if (wave == 0) {
            if (t + 1 < nT) {            // async-fetch next buffer, overlap with WMMA
                const int nb = buf ^ 1;
                tdm_load_tile((unsigned)(nb * 2 * TILE_H * sizeof(_Float16)),
                              Arow + (size_t)(t + 1) * BK, g1);
                tdm_load_tile((unsigned)((nb * 2 + 1) * TILE_H * sizeof(_Float16)),
                              Wrow + (size_t)(t + 1) * BK, g1);
                __builtin_amdgcn_s_wait_tensorcnt((short)2);  // current buf done
            } else {
                __builtin_amdgcn_s_wait_tensorcnt((short)0);
            }
        }
        __syncthreads();
        // TDM writes LDS behind the compiler's back (descriptor carries only an
        // integer LDS offset). Escape `sh` + memory clobber so the ds_loads
        // below must actually read LDS and cannot be folded to undef.
        asm volatile("" : : "v"((const void*)sh) : "memory");
#else
        // Fallback staging: global -> VGPR -> LDS
#pragma unroll
        for (int i = 0; i < 2; ++i) {
            const int vi  = tid + 256 * i;
            const int row = vi >> 2;
            const int kc  = (vi & 3) * 8;
            v8h av = *(const v8h*)(Arow + (size_t)row * kDIn + t * BK + kc);
            v8h wv = *(const v8h*)(Wrow + (size_t)row * kDIn + t * BK + kc);
            *(v8h*)((_Float16*)Ab + row * LDK + kc) = av;
            *(v8h*)((_Float16*)Wb + row * LDK + kc) = wv;
        }
        if (t + 1 < nT) {
            __builtin_prefetch(Arow + (size_t)(tid >> 1) * kDIn + (t + 1) * BK, 0, 1);
            __builtin_prefetch(Wrow + (size_t)(tid >> 1) * kDIn + (t + 1) * BK, 0, 1);
        }
        __syncthreads();
#endif

        // ---- fragments per ISA 7.12.2 16-bit layouts ----
        v16h a[2], b[4];
#pragma unroll
        for (int i = 0; i < 2; ++i) {
            // A lane(fr,hi): halves 0-7 = K 8*hi.., halves 8-15 = K 16+8*hi..
            const _Float16* p = Ab + (wm + i * 16 + fr) * LDK;
            v8h lo = *(const v8h*)(p + 8 * hi);
            v8h hh = *(const v8h*)(p + 16 + 8 * hi);
            a[i] = __builtin_shufflevector(lo, hh, 0, 1, 2, 3, 4, 5, 6, 7,
                                                   8, 9, 10, 11, 12, 13, 14, 15);
        }
#pragma unroll
        for (int j = 0; j < 4; ++j) {
            // B lane(fr,hi): halves 0-7 = K 8*hi.., halves 8-15 = K 16+8*hi.. of weight row n = fr
            const _Float16* p = Wb + (wn + j * 16 + fr) * LDK;
            v8h lo = *(const v8h*)(p + 8 * hi);
            v8h hh = *(const v8h*)(p + 16 + 8 * hi);
            b[j] = __builtin_shufflevector(lo, hh, 0, 1, 2, 3, 4, 5, 6, 7,
                                                   8, 9, 10, 11, 12, 13, 14, 15);
        }

#pragma unroll
        for (int i = 0; i < 2; ++i)
#pragma unroll
            for (int j = 0; j < 4; ++j)
                acc[i][j] = __builtin_amdgcn_wmma_f32_16x16x32_f16(
                    false, a[i], false, b[j], (short)0, acc[i][j], false, false);

        __syncthreads();   // all waves done with buf before it is refilled
    }

    // ---- epilogue: D layout VGPR v, lane: M = v + 8*hi, N = fr ----
#pragma unroll
    for (int j = 0; j < 4; ++j) {
        const int col = n_blk + wn + j * 16 + fr;
        const float bv = bias[col];
#pragma unroll
        for (int i = 0; i < 2; ++i) {
            const int row0 = m_blk + wm + i * 16 + 8 * hi;
#pragma unroll
            for (int v = 0; v < 8; ++v) {
                __builtin_nontemporal_store(acc[i][j][v] + bv,
                    Out + (size_t)(row0 + v) * kDOut + col);
            }
        }
    }
}

extern "C" void kernel_launch(void* const* d_in, const int* in_sizes, int n_in,
                              void* d_out, int out_size, void* d_ws, size_t ws_size,
                              hipStream_t stream) {
    const float* X     = (const float*)d_in[0];
    const int*   Q     = (const int*)d_in[1];
    const float* scale = (const float*)d_in[2];
    const float* zp    = (const float*)d_in[3];
    const float* bias  = (const float*)d_in[4];
    float* Out = (float*)d_out;

    _Float16* Xh = (_Float16*)d_ws;            // 128 MB
    _Float16* Wh = Xh + kElems;                // 128 MB

    const int cvt_blocks = (int)(kElems / (256 * 8));
    cvt_dequant_kernel<<<cvt_blocks, 256, 0, stream>>>(X, Q, scale, zp, Xh, Wh);

    dim3 grid(kDOut / BN, kRows / BM);
    qlinear_wmma_kernel<<<grid, 256, 0, stream>>>(Xh, Wh, bias, Out);
}
